// deformconv2d_47115791237774
// MI455X (gfx1250) — compile-verified, matrix operand order checked by reading, not run on hardware
//
#include <hip/hip_runtime.h>
#include <hip/hip_bf16.h>
#include <stdint.h>

// ---------------- problem constants (from reference) ----------------
#define Bsz   8
#define Cch   128
#define Hh    64
#define Ww    64
#define Oo    128
#define KTAP  9            // 3x3
#define KDIM  (Cch * KTAP) // 1152 = GEMM K
#define NKCH  (KDIM / 32)  // 36 chunks of K=32 (bf16 WMMA)
#define HW    (Hh * Ww)    // 4096
#define NPIX  (Bsz * HW)   // 32768 output pixels = GEMM N
#define TILE_N 32          // pixels per workgroup = two WMMA N-tiles
#define NTASK (TILE_N * KTAP) // 288 (pixel,tap) sampling tasks per WG

typedef __attribute__((ext_vector_type(16))) __bf16 v16bf;
typedef __attribute__((ext_vector_type(8)))  float  v8f;

static __device__ __forceinline__ unsigned short f2bf(float f) {
    union { float f; uint32_t u; } v; v.f = f;
    uint32_t u = v.u;
    uint32_t r = u + 0x7FFFu + ((u >> 16) & 1u);   // round-to-nearest-even
    return (unsigned short)(r >> 16);
}
static __device__ __forceinline__ float bf2f(unsigned short h) {
    union { float f; uint32_t u; } v; v.u = ((uint32_t)h) << 16;  // exact
    return v.f;
}

// ------------- kernel 0a: (B,C,H,W) f32 -> (B,H,W,C) bf16 tiled transpose ---
__global__ __launch_bounds__(256) void transpose_bhwc_kernel(
        const float* __restrict__ in, unsigned short* __restrict__ bhwc) {
    __shared__ float tile[32][33];                 // +1 pad kills bank conflicts
    const int s0 = blockIdx.x * 32;                // spatial (y*W+x) tile base
    const int c0 = blockIdx.y * 32;                // channel tile base
    const int b  = blockIdx.z;
    const int tx = threadIdx.x, ty = threadIdx.y;  // 32 x 8
    #pragma unroll
    for (int i = 0; i < 4; ++i) {                  // coalesced along spatial
        const int c = c0 + ty + 8 * i;
        tile[ty + 8 * i][tx] = in[((size_t)(b * Cch + c)) * HW + s0 + tx];
    }
    __syncthreads();
    #pragma unroll
    for (int i = 0; i < 4; ++i) {                  // coalesced along channel
        const int s = s0 + ty + 8 * i;
        bhwc[((size_t)b * HW + s) * Cch + c0 + tx] = f2bf(tile[tx][ty + 8 * i]);
    }
}

// ------------- kernel 0b: pack fp32 weights -> bf16 WMMA A-fragments --------
// Layout: wp[((mtile*NKCH + kchunk)*32 + lane)*16 + half]
__global__ __launch_bounds__(256) void pack_weights_kernel(
        const float* __restrict__ w, unsigned short* __restrict__ wp) {
    int e = blockIdx.x * blockDim.x + threadIdx.x;
    const int total = 8 * NKCH * 32 * 16;          // 147456 bf16 elements
    if (e >= total) return;
    int half   = e & 15;
    int lane   = (e >> 4) & 31;
    int kchunk = (e >> 9) % NKCH;
    int mtile  = e / (NKCH * 512);
    int o  = mtile * 16 + (lane & 15);             // A row M = lane%16
    // 16-bit fragment layout (CDNA5 ISA): lanes 0-15 hold K 0-7,16-23;
    // lanes 16-31 hold K 8-15,24-31.
    int kc = (half & 7) + ((half >> 3) * 16) + ((lane >> 4) * 8);
    int k  = kchunk * 32 + kc;                     // k = tap*128 + c
    int tap = k >> 7;
    int c   = k & (Cch - 1);
    wp[e] = f2bf(w[(o * Cch + c) * KTAP + tap]);
}

// ------------- kernel 1: deformable sampling + dual-N WMMA GEMM -------------
__global__ __launch_bounds__(256) void deform_conv_wmma_kernel(
        const unsigned short* __restrict__ bhwc,  // (B,H,W,C) bf16
        const float* __restrict__ off,            // (B,18,H,W)
        const float* __restrict__ msk,            // (B,9,H,W)
        const unsigned short* __restrict__ wpack, // packed bf16 A fragments
        const float* __restrict__ bias,           // (O)
        float* __restrict__ out)                  // (B,O,H,W)
{
    // Two N-tiles of B panel: 2 x 36 chunks x (32 lanes x 16 halves) = 72 KB
    __shared__ alignas(32) unsigned short Bp[2 * NKCH * 512];
    // per-(pixel,tap) bilinear params: corner element offsets + weights
    __shared__ int   sa00[NTASK], sa01[NTASK], sa10[NTASK], sa11[NTASK];
    __shared__ float sw00[NTASK], sw01[NTASK], sw10[NTASK], sw11[NTASK];

    const int tid     = threadIdx.x;
    const int pixbase = blockIdx.x * TILE_N;       // 32 pixels = one half-row
    const int b       = pixbase >> 12;             // / HW
    const int oyox    = pixbase & (HW - 1);
    const int oy      = oyox >> 6;                 // / Ww
    const int oxbase  = oyox & (Ww - 1);

    // ---- Phase A: bilinear parameters, 288 tasks over 256 threads
    for (int i = tid; i < NTASK; i += 256) {
        const int pix = i & 31;
        const int tap = i >> 5;                    // 0..8
        const int ox  = oxbase + pix;
        const int kh  = tap / 3, kw = tap - 3 * kh;
        const int sp  = oy * Ww + ox;
        const float offy = off[((size_t)(b * 18 + tap * 2 + 0)) * HW + sp];
        const float offx = off[((size_t)(b * 18 + tap * 2 + 1)) * HW + sp];
        const float m    = msk[((size_t)(b * KTAP + tap)) * HW + sp];
        const float py = (float)(oy - 1 + kh) + offy;   // STRIDE=1,PAD=1,DIL=1
        const float px = (float)(ox - 1 + kw) + offx;
        const float y0f = floorf(py), x0f = floorf(px);
        const int   iy0 = (int)y0f,  ix0 = (int)x0f;
        const float wy1 = py - y0f,  wx1 = px - x0f;
        const float wy0 = 1.0f - wy1, wx0 = 1.0f - wx1;
        const bool vy0 = (iy0 >= 0) & (iy0 < Hh);
        const bool vy1 = (iy0 + 1 >= 0) & (iy0 + 1 < Hh);
        const bool vx0 = (ix0 >= 0) & (ix0 < Ww);
        const bool vx1 = (ix0 + 1 >= 0) & (ix0 + 1 < Ww);
        const int y0c = min(max(iy0, 0), Hh - 1),  y1c = min(max(iy0 + 1, 0), Hh - 1);
        const int x0c = min(max(ix0, 0), Ww - 1),  x1c = min(max(ix0 + 1, 0), Ww - 1);
        sa00[i] = (y0c * Ww + x0c) * Cch;          // element offsets into bhwc
        sa01[i] = (y0c * Ww + x1c) * Cch;
        sa10[i] = (y1c * Ww + x0c) * Cch;
        sa11[i] = (y1c * Ww + x1c) * Cch;
        sw00[i] = wy0 * wx0 * ((vy0 & vx0) ? m : 0.0f);
        sw01[i] = wy0 * wx1 * ((vy0 & vx1) ? m : 0.0f);
        sw10[i] = wy1 * wx0 * ((vy1 & vx0) ? m : 0.0f);
        sw11[i] = wy1 * wx1 * ((vy1 & vx1) ? m : 0.0f);
    }
    __syncthreads();

    // ---- Phase B: one wave per (pixel,tap) task; lane = 4 channels.
    // Each corner row is 256 B contiguous bf16 -> fully coalesced 8 B/lane.
    {
        const int wave = tid >> 5;
        const int lane = tid & 31;
        const int c0   = lane * 4;
        const unsigned short* __restrict__ ibh = bhwc + (size_t)b * HW * Cch;
        for (int t = wave; t < NTASK; t += 8) {
            const int pix = t & 31;
            const int tap = t >> 5;
            const float w00 = sw00[t], w01 = sw01[t];
            const float w10 = sw10[t], w11 = sw11[t];
            const ushort4 q00 = *(const ushort4*)(ibh + sa00[t] + c0);
            const ushort4 q01 = *(const ushort4*)(ibh + sa01[t] + c0);
            const ushort4 q10 = *(const ushort4*)(ibh + sa10[t] + c0);
            const ushort4 q11 = *(const ushort4*)(ibh + sa11[t] + c0);
            unsigned short h[4];
            h[0] = f2bf(bf2f(q00.x)*w00 + bf2f(q01.x)*w01 + bf2f(q10.x)*w10 + bf2f(q11.x)*w11);
            h[1] = f2bf(bf2f(q00.y)*w00 + bf2f(q01.y)*w01 + bf2f(q10.y)*w10 + bf2f(q11.y)*w11);
            h[2] = f2bf(bf2f(q00.z)*w00 + bf2f(q01.z)*w01 + bf2f(q10.z)*w10 + bf2f(q11.z)*w11);
            h[3] = f2bf(bf2f(q00.w)*w00 + bf2f(q01.w)*w01 + bf2f(q10.w)*w10 + bf2f(q11.w)*w11);
            // B-fragment address: kIdx = tap*128 + c0 (+0..3 -> contiguous halves)
            const int kIdx   = tap * Cch + c0;
            const int kchunk = kIdx >> 5;
            const int kc     = kIdx & 31;
            const int lanegrp  = (kc >> 3) & 1;
            const int halfbase = (kc & 7) | (((kc >> 4) & 1) << 3);
            const int ntile = pix >> 4, nn = pix & 15;
            unsigned short* dst =
                &Bp[(((ntile * NKCH + kchunk) * 32) + nn + (lanegrp << 4)) * 16 + halfbase];
            uint2 pk; pk.x = (uint32_t)h[0] | ((uint32_t)h[1] << 16);
            pk.y = (uint32_t)h[2] | ((uint32_t)h[3] << 16);
            *(uint2*)dst = pk;                     // 8 B aligned store
        }
    }
    __syncthreads();

    // ---- Phase C: 8 waves x one 16-row M-tile x two N-tiles, K=1152
    {
        const int wave = tid >> 5;                 // = M-tile 0..7
        const int lane = tid & 31;
        const v16bf* __restrict__ Afr =
            (const v16bf*)wpack + (size_t)wave * NKCH * 32 + lane;
        v8f acc0 = {0.f,0.f,0.f,0.f,0.f,0.f,0.f,0.f};
        v8f acc1 = {0.f,0.f,0.f,0.f,0.f,0.f,0.f,0.f};
        #pragma unroll 4
        for (int kchunk = 0; kchunk < NKCH; ++kchunk) {
            if (kchunk + 1 < NKCH)                 // warm next A chunk
                __builtin_prefetch((const void*)(Afr + (size_t)(kchunk + 1) * 32), 0, 1);
            const v16bf a  = Afr[(size_t)kchunk * 32];
            const v16bf b0 = *(const v16bf*)&Bp[(kchunk * 32 + lane) * 16];
            const v16bf b1 = *(const v16bf*)&Bp[((NKCH + kchunk) * 32 + lane) * 16];
            acc0 = __builtin_amdgcn_wmma_f32_16x16x32_bf16(
                       false, a, false, b0, (short)0, acc0, false, false);
            acc1 = __builtin_amdgcn_wmma_f32_16x16x32_bf16(
                       false, a, false, b1, (short)0, acc1, false, false);
        }
        // C/D layout: lanes 0-15 -> M=r, lanes 16-31 -> M=8+r; N = lane%16
        const int n    = lane & 15;
        const int row8 = (lane >> 4) * 8;
        float* __restrict__ ob = out + ((size_t)b * Oo) * HW + oyox;
        #pragma unroll
        for (int r = 0; r < 8; ++r) {
            const int o = wave * 16 + row8 + r;
            const float bo = bias[o];
            ob[(size_t)o * HW + n]      = acc0[r] + bo;
            ob[(size_t)o * HW + 16 + n] = acc1[r] + bo;
        }
    }
}

// ---------------- host-side launch ----------------
extern "C" void kernel_launch(void* const* d_in, const int* in_sizes, int n_in,
                              void* d_out, int out_size, void* d_ws, size_t ws_size,
                              hipStream_t stream) {
    const float* inp  = (const float*)d_in[0];   // (8,128,64,64)
    const float* off  = (const float*)d_in[1];   // (8,18,64,64)
    const float* msk  = (const float*)d_in[2];   // (8,9,64,64)
    const float* wgt  = (const float*)d_in[3];   // (128,128,3,3)
    const float* bias = (const float*)d_in[4];   // (128,)
    float* out = (float*)d_out;

    // workspace: [0, 8 MB) BHWC bf16 input; then 288 KB packed weights
    unsigned short* bhwc  = (unsigned short*)d_ws;
    unsigned short* wpack = (unsigned short*)((char*)d_ws
                              + (size_t)Bsz * HW * Cch * sizeof(unsigned short));

    transpose_bhwc_kernel<<<dim3(HW / 32, Cch / 32, Bsz), dim3(32, 8), 0, stream>>>(
        inp, bhwc);

    const int packElems = 8 * NKCH * 32 * 16;      // 147456
    pack_weights_kernel<<<(packElems + 255) / 256, 256, 0, stream>>>(wgt, wpack);

    const int nBlocks = NPIX / TILE_N;             // 1024
    deform_conv_wmma_kernel<<<nBlocks, 256, 0, stream>>>(
        bhwc, off, msk, wpack, bias, out);
}
